// PointTransformerLayer_L_78554951844346
// MI455X (gfx1250) — matrix operand order repaired and checked by reading, not compiled, not run on hardware
//
#include <hip/hip_runtime.h>
#include <hip/hip_bf16.h>

// ---------------------------------------------------------------------------
// Point Transformer layer for MI455X (gfx1250, wave32, WMMA).
// All GEMMs (incl. final fc) run as v_wmma_f32_16x16x32_bf16 (f32 accumulate).
// Weights pre-packed to bf16 B-operand lane layout in d_ws by a prologue
// kernel. One wave per point. f32 LDS only where precision matters
// (residual + LN); all WMMA-feeding intermediates staged in bf16 LDS so
// fragment packing is raw ds_load_b128 (no per-read conversion VALU).
// ---------------------------------------------------------------------------

#define BB 4
#define NN 8192
#define CC 128
#define NS 16
#define HH 4
#define DH 32

// packed-weight offsets in d_ws (bf16 units). tile = 512 elements.
#define OFF_INPROJ 0        // 384x128 -> 24*4 tiles = 49152
#define OFF_OUT    49152    // 128x128 -> 8*4 tiles  = 16384
#define OFF_FF1    65536    // 256x128 -> 16*4 tiles = 32768
#define OFF_FF2    98304    // 128x256 -> 8*8 tiles  = 32768
#define OFF_FC     131072   // 128x128 -> 8*4 tiles  = 16384
#define WPK_TOTAL  147456
#define POFF (BB * NN * 3)  // p passthrough size in d_out

typedef __attribute__((ext_vector_type(16))) __bf16 v16bf;
typedef __attribute__((ext_vector_type(16))) float  v16f;
typedef __attribute__((ext_vector_type(8)))  float  v8f;

union V16 {
  v16bf v;
  __bf16 e[16];
  unsigned int w[8];
};

__device__ __forceinline__ v8f wmma_bf16(v16bf a, v16bf b, v8f c) {
  return __builtin_amdgcn_wmma_f32_16x16x32_bf16(
      /*neg_a=*/false, a, /*neg_b=*/false, b,
      /*c_mod=*/(short)0, c, /*reuse_a=*/false, /*reuse_b=*/false);
}

// Assemble a v16bf fragment from two 16-byte chunks of bf16 data.
__device__ __forceinline__ v16bf frag2x16B(const __bf16* p0, const __bf16* p1) {
  uint4 a = *(const uint4*)p0;
  uint4 b = *(const uint4*)p1;
  V16 u;
  u.w[0] = a.x; u.w[1] = a.y; u.w[2] = a.z; u.w[3] = a.w;
  u.w[4] = b.x; u.w[5] = b.y; u.w[6] = b.z; u.w[7] = b.w;
  return u.v;
}

// ---- A fragment layouts ---------------------------------------------------
// 16-bit A layout: lane m=lane&15, half=lane>>4;
//   K = colBase + half*16 + j
// ---------------------------------------------------------------------------

// A fragment from bf16 LDS (row stride rs elements): pure ds_load_b128 x2.
__device__ __forceinline__ v16bf pack_A16(const __bf16* base, int rs, int colBase, int lane) {
  int m = lane & 15, half = lane >> 4;
  const __bf16* r = base + m * rs + colBase + half * 16;
  return frag2x16B(r, r + 8);
}

// A fragment from f32 LDS: 4x ds_load_b128 + packed cvt to bf16.
// rs==0 broadcasts one row into all 16 A rows (used for the fc matvec).
__device__ __forceinline__ v16bf pack_Af32(const float* base, int rs, int colBase, int lane) {
  int m = lane & 15, half = lane >> 4;
  const float* r = base + m * rs + colBase + half * 16;
  float4 a = *(const float4*)(r);
  float4 b = *(const float4*)(r + 4);
  float4 c = *(const float4*)(r + 8);
  float4 d = *(const float4*)(r + 12);
  v16f t;
  t[0] = a.x;  t[1] = a.y;  t[2]  = a.z;  t[3]  = a.w;
  t[4] = b.x;  t[5] = b.y;  t[6]  = b.z;  t[7]  = b.w;
  t[8] = c.x;  t[9] = c.y;  t[10] = c.z;  t[11] = c.w;
  t[12] = d.x; t[13] = d.y; t[14] = d.z;  t[15] = d.w;
  return __builtin_convertvector(t, v16bf);
}

// attn A fragment from bf16 LDS (16x16), zero-padded to K=32.
__device__ __forceinline__ v16bf pack_A_attn16(const __bf16* att, int lane) {
  int m = lane & 15, half = lane >> 4;
  uint4 a = *(const uint4*)(att + m * 16);
  uint4 b = *(const uint4*)(att + m * 16 + 8);
  V16 u;
  u.w[0] = half ? 0u : a.x; u.w[1] = half ? 0u : a.y; u.w[2] = half ? 0u : a.z; u.w[3] = half ? 0u : a.w;
  u.w[4] = half ? 0u : b.x; u.w[5] = half ? 0u : b.y; u.w[6] = half ? 0u : b.z; u.w[7] = half ? 0u : b.w;   // K=16..31 (half 1) -> 0
  return u.v;
}

// ---- B fragment layouts ---------------------------------------------------
// B layout: lane n=lane&15, half=lane>>4; K(j) = half*16 + j.
// ---------------------------------------------------------------------------

// B[K][n] = data[n][colBase+K] with bf16 LDS rows (K contiguous per row n).
__device__ __forceinline__ v16bf pack_B16_rowT(const __bf16* base, int rs, int colBase, int lane) {
  int n = lane & 15, half = lane >> 4;
  const __bf16* r = base + n * rs + colBase + half * 16;
  return frag2x16B(r, r + 8);
}

// B[K][n] = base[K*rs + colBase + n], valid only for K<16 (upper half zeroed;
// paired with the zero-padded attn A fragment).
__device__ __forceinline__ v16bf pack_B16_col(const __bf16* base, int rs, int colBase, int lane) {
  int n = lane & 15, half = lane >> 4;
  V16 u;
#pragma unroll
  for (int j = 0; j < 16; ++j)
    u.e[j] = (half == 0) ? base[j * rs + colBase + n] : (__bf16)0.0f;
  return u.v;
}

// Pre-packed bf16 weight tile: 32 lanes x 16 bf16 contiguous (32B/lane).
__device__ __forceinline__ v16bf load_B_w(const __bf16* w, int tile, int lane) {
  const __bf16* r = w + ((tile * 32 + lane) << 4);
  return frag2x16B(r, r + 8);
}

__device__ __forceinline__ void layernorm16(float (*F)[CC], const float* g,
                                            const float* b, int lane) {
  float4 g4 = *(const float4*)(g + lane * 4);
  float4 b4 = *(const float4*)(b + lane * 4);
  for (int s = 0; s < NS; ++s) {
    float4 xv = *(const float4*)(&F[s][lane * 4]);
    float sum = xv.x + xv.y + xv.z + xv.w;
    float sq  = xv.x * xv.x + xv.y * xv.y + xv.z * xv.z + xv.w * xv.w;
#pragma unroll
    for (int m = 1; m < 32; m <<= 1) {
      sum += __shfl_xor(sum, m, 32);
      sq  += __shfl_xor(sq, m, 32);
    }
    float mean = sum * (1.0f / 128.0f);
    float var  = sq * (1.0f / 128.0f) - mean * mean;
    float rs   = rsqrtf(var + 1e-5f);
    float4 o;
    o.x = (xv.x - mean) * rs * g4.x + b4.x;
    o.y = (xv.y - mean) * rs * g4.y + b4.y;
    o.z = (xv.z - mean) * rs * g4.z + b4.z;
    o.w = (xv.w - mean) * rs * g4.w + b4.w;
    *(float4*)(&F[s][lane * 4]) = o;
  }
}

// ---------------------------------------------------------------------------
// Prologue: pack f32 weights -> bf16 WMMA-B lane layout in d_ws.
// Packed element e -> tile t = e/512, lane = (e%512)/16, j = e%16.
// B[K][n] = W[n][k]; n = nt*16 + (lane&15); k = kt*32 + (lane>>4)*16 + j.
// ---------------------------------------------------------------------------
__global__ __launch_bounds__(256)
void pack_weights(const float* __restrict__ inproj, const float* __restrict__ outw,
                  const float* __restrict__ ff1w,  const float* __restrict__ ff2w,
                  const float* __restrict__ fcw,   __bf16* __restrict__ ws) {
  int e = blockIdx.x * 256 + threadIdx.x;
  if (e >= WPK_TOTAL) return;
  const float* W; int Cin; int rel;
  if (e < OFF_OUT)      { W = inproj; Cin = 128; rel = e; }
  else if (e < OFF_FF1) { W = outw;   Cin = 128; rel = e - OFF_OUT; }
  else if (e < OFF_FF2) { W = ff1w;   Cin = 128; rel = e - OFF_FF1; }
  else if (e < OFF_FC)  { W = ff2w;   Cin = 256; rel = e - OFF_FF2; }
  else                  { W = fcw;    Cin = 128; rel = e - OFF_FC; }
  int t = rel >> 9, within = rel & 511;
  int lane = within >> 4, j = within & 15;
  int KT = Cin >> 5;
  int nt = t / KT, kt = t - nt * KT;
  int row = nt * 16 + (lane & 15);
  int k   = kt * 32 + (lane >> 4) * 16 + j;
  ws[e] = (__bf16)W[row * Cin + k];
}

// ---------------------------------------------------------------------------
// Main kernel: one wave32 per point.
// ---------------------------------------------------------------------------
__global__ __launch_bounds__(32)
void pt_layer(const float* __restrict__ p,  const float* __restrict__ x,
              const int* __restrict__ idx,
              const float* __restrict__ pe_w1, const float* __restrict__ bn_g,
              const float* __restrict__ bn_b,  const float* __restrict__ bn_m,
              const float* __restrict__ bn_v,  const float* __restrict__ pe_w2,
              const float* __restrict__ pe_b2,
              const float* __restrict__ in_proj_b, const float* __restrict__ out_b,
              const float* __restrict__ ln1_g, const float* __restrict__ ln1_b,
              const float* __restrict__ ff1_b, const float* __restrict__ ff2_b,
              const float* __restrict__ ln2_g, const float* __restrict__ ln2_b,
              const float* __restrict__ fc_b,
              const __bf16* __restrict__ wpk,
              float* __restrict__ outp) {
  __shared__ float  lds_f[NS][CC];       // feature / residual, f32 (8KB)
  __shared__ __bf16 lds_t[NS][3 * CC];   // qkv / ffn hidden, bf16 (12KB)
  __shared__ __bf16 lds_o[NS][CC];       // attention output, bf16 (4KB)
  __shared__ __bf16 lds_att[NS][NS];     // attn weights, bf16 (512B)
  __shared__ float  lds_pool[CC];        // pooled features (512B)
  __shared__ float  lds_t3[3][NS];       // PE intermediate
  __shared__ int    lds_idx[NS];

  const int P    = blockIdx.x;
  const int bb   = P >> 13;        // N = 8192
  const int n    = P & (NN - 1);
  const int lane = threadIdx.x;
  const int col  = lane & 15;
  const int half = lane >> 4;

  // ---- pass-through of p into d_out[0 .. B*N*3) ----
  if (lane < 3) outp[(size_t)P * 3 + lane] = p[(size_t)P * 3 + lane];

  // ---- stage 1: neighbor gather + positional encoding ----
  if (lane < NS) {
    int s  = lane;
    int id = idx[((size_t)bb * NN + n) * NS + s];
    lds_idx[s] = id;
    float r0 = p[((size_t)bb * NN + id) * 3 + 0] - p[((size_t)bb * NN + n) * 3 + 0];
    float r1 = p[((size_t)bb * NN + id) * 3 + 1] - p[((size_t)bb * NN + n) * 3 + 1];
    float r2 = p[((size_t)bb * NN + id) * 3 + 2] - p[((size_t)bb * NN + n) * 3 + 2];
#pragma unroll
    for (int o = 0; o < 3; ++o) {
      float t = pe_w1[o * 3 + 0] * r0 + pe_w1[o * 3 + 1] * r1 + pe_w1[o * 3 + 2] * r2;
      t = (t - bn_m[o]) * rsqrtf(bn_v[o] + 1e-5f);
      t = t * bn_g[o] + bn_b[o];
      lds_t3[o][s] = fmaxf(t, 0.0f);
    }
  }
  __syncthreads();
#pragma unroll 4
  for (int i = 0; i < 64; ++i) {
    int e = lane + 32 * i;
    int s = e & 15, c = e >> 4;
    float gx = x[(((size_t)bb * CC + c) << 13) + lds_idx[s]];
    float pe = pe_w2[c * 3 + 0] * lds_t3[0][s] +
               pe_w2[c * 3 + 1] * lds_t3[1][s] +
               pe_w2[c * 3 + 2] * lds_t3[2][s] + pe_b2[c];
    lds_f[s][c] = gx + pe;
  }
  __syncthreads();

  // ---- stage 2: qkv = f @ in_proj_w^T + b  (M=16, K=128, N=384) ----
  {
    v16bf af[4];
#pragma unroll
    for (int kt = 0; kt < 4; ++kt) af[kt] = pack_Af32(&lds_f[0][0], CC, kt * 32, lane);
    for (int nt = 0; nt < 24; ++nt) {
      v8f acc = {};
#pragma unroll
      for (int kt = 0; kt < 4; ++kt)
        acc = wmma_bf16(af[kt], load_B_w(wpk + OFF_INPROJ, nt * 4 + kt, lane), acc);
      float bias = in_proj_b[nt * 16 + col];
#pragma unroll
      for (int r = 0; r < 8; ++r)
        lds_t[r + 8 * half][nt * 16 + col] = (__bf16)(acc[r] + bias);
    }
  }
  __syncthreads();

  // ---- stage 3: attention (4 heads, dh=32, 16 keys) ----
  for (int h = 0; h < HH; ++h) {
    v16bf qf = pack_A16(&lds_t[0][0], 3 * CC, h * 32, lane);
    v16bf kf = pack_B16_rowT(&lds_t[0][0], 3 * CC, CC + h * 32, lane);
    v8f sc = {};
    sc = wmma_bf16(qf, kf, sc);
#pragma unroll
    for (int r = 0; r < 8; ++r) {
      float v = sc[r] * 0.17677669529663687f;   // 1/sqrt(32)
      float mx = v;
#pragma unroll
      for (int m = 1; m < 16; m <<= 1) mx = fmaxf(mx, __shfl_xor(mx, m, 32));
      float e = __expf(v - mx);
      float s = e;
#pragma unroll
      for (int m = 1; m < 16; m <<= 1) s += __shfl_xor(s, m, 32);
      lds_att[r + 8 * half][col] = (__bf16)(e / s);
    }
    __syncthreads();
    v16bf af = pack_A_attn16(&lds_att[0][0], lane);
#pragma unroll
    for (int dt = 0; dt < 2; ++dt) {
      v16bf vf = pack_B16_col(&lds_t[0][0], 3 * CC, 2 * CC + h * 32 + dt * 16, lane);
      v8f oc = {};
      oc = wmma_bf16(af, vf, oc);
#pragma unroll
      for (int r = 0; r < 8; ++r)
        lds_o[r + 8 * half][h * 32 + dt * 16 + col] = (__bf16)oc[r];
    }
    __syncthreads();
  }

  // ---- stage 4: out_proj + residual + LN1 ----
  {
    v16bf af[4];
#pragma unroll
    for (int kt = 0; kt < 4; ++kt) af[kt] = pack_A16(&lds_o[0][0], CC, kt * 32, lane);
    for (int nt = 0; nt < 8; ++nt) {
      v8f acc = {};
#pragma unroll
      for (int kt = 0; kt < 4; ++kt)
        acc = wmma_bf16(af[kt], load_B_w(wpk + OFF_OUT, nt * 4 + kt, lane), acc);
      float bias = out_b[nt * 16 + col];
#pragma unroll
      for (int r = 0; r < 8; ++r)
        lds_f[r + 8 * half][nt * 16 + col] += acc[r] + bias;
    }
  }
  __syncthreads();
  layernorm16(lds_f, ln1_g, ln1_b, lane);
  __syncthreads();

  // ---- stage 5: ff1 (relu) ----
  {
    v16bf af[4];
#pragma unroll
    for (int kt = 0; kt < 4; ++kt) af[kt] = pack_Af32(&lds_f[0][0], CC, kt * 32, lane);
    for (int nt = 0; nt < 16; ++nt) {
      v8f acc = {};
#pragma unroll
      for (int kt = 0; kt < 4; ++kt)
        acc = wmma_bf16(af[kt], load_B_w(wpk + OFF_FF1, nt * 4 + kt, lane), acc);
      float bias = ff1_b[nt * 16 + col];
#pragma unroll
      for (int r = 0; r < 8; ++r)
        lds_t[r + 8 * half][nt * 16 + col] = (__bf16)fmaxf(acc[r] + bias, 0.0f);
    }
  }
  __syncthreads();

  // ---- stage 6: ff2 + residual + LN2 ----
  {
    v16bf af[8];
#pragma unroll
    for (int kt = 0; kt < 8; ++kt) af[kt] = pack_A16(&lds_t[0][0], 3 * CC, kt * 32, lane);
    for (int nt = 0; nt < 8; ++nt) {
      v8f acc = {};
#pragma unroll
      for (int kt = 0; kt < 8; ++kt)
        acc = wmma_bf16(af[kt], load_B_w(wpk + OFF_FF2, nt * 8 + kt, lane), acc);
      float bias = ff2_b[nt * 16 + col];
#pragma unroll
      for (int r = 0; r < 8; ++r)
        lds_f[r + 8 * half][nt * 16 + col] += acc[r] + bias;
    }
  }
  __syncthreads();
  layernorm16(lds_f, ln2_g, ln2_b, lane);
  __syncthreads();

  // ---- stage 7: max-pool over ns ----
  {
    float4 pm = *(const float4*)(&lds_f[0][lane * 4]);
    for (int s = 1; s < NS; ++s) {
      float4 xv = *(const float4*)(&lds_f[s][lane * 4]);
      pm.x = fmaxf(pm.x, xv.x); pm.y = fmaxf(pm.y, xv.y);
      pm.z = fmaxf(pm.z, xv.z); pm.w = fmaxf(pm.w, xv.w);
    }
    *(float4*)(&lds_pool[lane * 4]) = pm;
  }
  __syncthreads();

  // ---- stage 8: fc as WMMA matvec (pooled broadcast into all A rows) ----
  {
    v16bf af[4];
#pragma unroll
    for (int kt = 0; kt < 4; ++kt) af[kt] = pack_Af32(&lds_pool[0], 0, kt * 32, lane);
    for (int nt = 0; nt < 8; ++nt) {
      v8f acc = {};
#pragma unroll
      for (int kt = 0; kt < 4; ++kt)
        acc = wmma_bf16(af[kt], load_B_w(wpk + OFF_FC, nt * 4 + kt, lane), acc);
      // all 16 D rows identical; lanes 0-15 hold row M=0 in acc[0]
      if (half == 0) {
        int o = nt * 16 + col;
        outp[POFF + ((size_t)bb * CC + o) * NN + n] = acc[0] + fc_b[o];
      }
    }
  }
}

extern "C" void kernel_launch(void* const* d_in, const int* in_sizes, int n_in,
                              void* d_out, int out_size, void* d_ws, size_t ws_size,
                              hipStream_t stream) {
  const float* p         = (const float*)d_in[0];
  const float* x         = (const float*)d_in[1];
  const int*   idx       = (const int*)d_in[2];
  const float* pe_w1     = (const float*)d_in[3];
  const float* bn_g      = (const float*)d_in[4];
  const float* bn_b      = (const float*)d_in[5];
  const float* bn_m      = (const float*)d_in[6];
  const float* bn_v      = (const float*)d_in[7];
  const float* pe_w2     = (const float*)d_in[8];
  const float* pe_b2     = (const float*)d_in[9];
  const float* in_proj_w = (const float*)d_in[10];
  const float* in_proj_b = (const float*)d_in[11];
  const float* out_w     = (const float*)d_in[12];
  const float* out_b     = (const float*)d_in[13];
  const float* ln1_g     = (const float*)d_in[14];
  const float* ln1_b     = (const float*)d_in[15];
  const float* ff1_w     = (const float*)d_in[16];
  const float* ff1_b     = (const float*)d_in[17];
  const float* ff2_w     = (const float*)d_in[18];
  const float* ff2_b     = (const float*)d_in[19];
  const float* ln2_g     = (const float*)d_in[20];
  const float* ln2_b     = (const float*)d_in[21];
  const float* fc_w      = (const float*)d_in[22];
  const float* fc_b      = (const float*)d_in[23];

  __bf16* wpk = (__bf16*)d_ws;

  pack_weights<<<(WPK_TOTAL + 255) / 256, 256, 0, stream>>>(in_proj_w, out_w,
                                                            ff1_w, ff2_w, fc_w, wpk);

  pt_layer<<<BB * NN, 32, 0, stream>>>(
      p, x, idx, pe_w1, bn_g, bn_b, bn_m, bn_v, pe_w2, pe_b2,
      in_proj_b, out_b, ln1_g, ln1_b, ff1_b, ff2_b, ln2_g, ln2_b,
      fc_b, wpk, (float*)d_out);
}
